// GATv2ConvGraphGymLayer_84576495992842
// MI455X (gfx1250) — matrix operand order repaired and checked by reading, not compiled, not run on hardware
//
#include <hip/hip_runtime.h>

typedef __attribute__((ext_vector_type(16))) _Float16 v16h;
typedef __attribute__((ext_vector_type(8)))  _Float16 v8h;
typedef __attribute__((ext_vector_type(8)))  float    v8f;

#define HEADS 4
#define CDIM 32
#define HC 128
#define DIN 128
#define DOUT 32
#define NEG_SLOPE 0.2f

// Order-preserving float <-> uint mapping so segment-max can use atomicMax(u32).
__device__ __forceinline__ unsigned encodeOrderedF32(float f) {
  unsigned i = __float_as_uint(f);
  return (i & 0x80000000u) ? ~i : (i | 0x80000000u);
}
__device__ __forceinline__ float decodeOrderedF32(unsigned k) {
  unsigned i = (k & 0x80000000u) ? (k & 0x7FFFFFFFu) : ~k;
  return __uint_as_float(i);
}

// ---------- prep: transpose weights to [HC][DIN] and convert to f16 ----------
__global__ void k_prep_weights(const float* __restrict__ Wl, const float* __restrict__ Wr,
                               _Float16* __restrict__ WlT, _Float16* __restrict__ WrT) {
  int t = blockIdx.x * blockDim.x + threadIdx.x;
  if (t >= HC * DIN) return;
  int n = t >> 7, k = t & 127;
  WlT[n * DIN + k] = (_Float16)Wl[k * HC + n];
  WrT[n * DIN + k] = (_Float16)Wr[k * HC + n];
}

// ---------- init accumulators each call ----------
__global__ void k_init(float* __restrict__ acc, float* __restrict__ denom,
                       unsigned* __restrict__ mEnc, int accN, int segN) {
  int t = blockIdx.x * blockDim.x + threadIdx.x;
  if (t < accN) acc[t] = 0.f;
  if (t < segN) { denom[t] = 0.f; mEnc[t] = 0x007FFFFFu; }  // encoded -inf
}

// ---------- WMMA transform: xl = x@Wl, xr = x@Wr (f16 in, f32 accum) ----------
__global__ __launch_bounds__(128) void k_transform(
    const float* __restrict__ x, const _Float16* __restrict__ WlT,
    const _Float16* __restrict__ WrT, float* __restrict__ xl,
    float* __restrict__ xr, int nNodes) {
  __shared__ __align__(16) _Float16 xs[16 * DIN];  // 4 KB f16 tile of x
  int row0 = blockIdx.x * 16;
  for (int i = threadIdx.x; i < 16 * DIN; i += 128) {
    int r = i >> 7, c = i & 127;
    int gr = row0 + r;
    xs[i] = (gr < nNodes) ? (_Float16)x[gr * DIN + c] : (_Float16)0.f;
  }
  __syncthreads();

  int wave = threadIdx.x >> 5;
  int lane = threadIdx.x & 31;
  int hs = lane >> 4;       // lane half-group selector
  int ln = lane & 15;

  v8f acc[4] = {};          // 4 output tiles per wave (16 tiles / 4 waves)
#pragma unroll
  for (int ks = 0; ks < 4; ++ks) {          // K = 128 = 4 x 32
    // A fragment (16x32 f16): lane holds row m=ln, K = 16h..16h+15
    int ka = ks * 32 + hs * 16;
    v16h a = *(const v16h*)(xs + ln * DIN + ka);

    int kb = ks * 32 + hs * 16;             // B: lanes 0-15 K=0..15, 16-31 K=16..31
#pragma unroll
    for (int t = 0; t < 4; ++t) {
      int tile = wave * 4 + t;              // 0..7 -> Wl tiles, 8..15 -> Wr tiles
      const _Float16* WT = (tile < 8) ? WlT : WrT;
      int nt = tile & 7;
      v16h b = *(const v16h*)(WT + (nt * 16 + ln) * DIN + kb);
      acc[t] = __builtin_amdgcn_wmma_f32_16x16x32_f16(
          false, a, false, b, (short)0, acc[t], false, false);
    }
  }

#pragma unroll
  for (int t = 0; t < 4; ++t) {
    int tile = wave * 4 + t;
    float* dstp = (tile < 8) ? xl : xr;
    int ncol = (tile & 7) * 16 + ln;
#pragma unroll
    for (int r = 0; r < 8; ++r) {           // C/D layout: VGPR r -> M = r + 8*hs
      int gr = row0 + r + hs * 8;
      if (gr < nNodes) dstp[gr * HC + ncol] = acc[t][r];
    }
  }
}

// ---------- per-edge attention logits + segment max ----------
__global__ void k_logits(const int* __restrict__ src, const int* __restrict__ dst,
                         const float* __restrict__ xl, const float* __restrict__ xr,
                         const float* __restrict__ att, float* __restrict__ logits,
                         unsigned* __restrict__ mEnc, int nEdges) {
  int e = blockIdx.x * blockDim.x + threadIdx.x;
  if (e >= nEdges) return;
  int js = src[e], it = dst[e];
  const float4* pl = (const float4*)(xl + (size_t)js * HC);
  const float4* pr = (const float4*)(xr + (size_t)it * HC);
  float lg[HEADS] = {0.f, 0.f, 0.f, 0.f};
#pragma unroll
  for (int q = 0; q < HC / 4; ++q) {
    float4 a4 = pl[q], b4 = pr[q];
    int h = q >> 3;
    const float* ah = att + h * CDIM + (q & 7) * 4;
    float v;
    v = a4.x + b4.x; v = v > 0.f ? v : v * NEG_SLOPE; lg[h] = fmaf(v, ah[0], lg[h]);
    v = a4.y + b4.y; v = v > 0.f ? v : v * NEG_SLOPE; lg[h] = fmaf(v, ah[1], lg[h]);
    v = a4.z + b4.z; v = v > 0.f ? v : v * NEG_SLOPE; lg[h] = fmaf(v, ah[2], lg[h]);
    v = a4.w + b4.w; v = v > 0.f ? v : v * NEG_SLOPE; lg[h] = fmaf(v, ah[3], lg[h]);
  }
#pragma unroll
  for (int h = 0; h < HEADS; ++h) {
    logits[(size_t)e * HEADS + h] = lg[h];
    atomicMax(&mEnc[(size_t)it * HEADS + h], encodeOrderedF32(lg[h]));
  }
}

// ---------- exp(logit - max) in place + segment sum ----------
__global__ void k_expsum(const int* __restrict__ dst, float* __restrict__ logits,
                         const unsigned* __restrict__ mEnc, float* __restrict__ denom,
                         int nEdges) {
  int e = blockIdx.x * blockDim.x + threadIdx.x;
  if (e >= nEdges) return;
  int it = dst[e];
#pragma unroll
  for (int h = 0; h < HEADS; ++h) {
    float mv = decodeOrderedF32(mEnc[(size_t)it * HEADS + h]);
    float ex = __expf(logits[(size_t)e * HEADS + h] - mv);
    logits[(size_t)e * HEADS + h] = ex;
    atomicAdd(&denom[(size_t)it * HEADS + h], ex);
  }
}

// ---------- weighted message scatter: 4 threads per edge (one per head) ----------
__global__ void k_scatter(const int* __restrict__ src, const int* __restrict__ dst,
                          const float* __restrict__ xl, const float* __restrict__ ex,
                          const float* __restrict__ denom, float* __restrict__ acc,
                          int nEdges) {
  int t = blockIdx.x * blockDim.x + threadIdx.x;
  int e = t >> 2, h = t & 3;
  if (e >= nEdges) return;
  int js = src[e], it = dst[e];
  float alpha = ex[(size_t)e * HEADS + h] / (denom[(size_t)it * HEADS + h] + 1e-16f);
  const float4* pl = (const float4*)(xl + (size_t)js * HC + h * CDIM);
  float* pa = acc + (size_t)it * HC + h * CDIM;
#pragma unroll
  for (int q = 0; q < CDIM / 4; ++q) {
    float4 v = pl[q];
    atomicAdd(pa + q * 4 + 0, v.x * alpha);
    atomicAdd(pa + q * 4 + 1, v.y * alpha);
    atomicAdd(pa + q * 4 + 2, v.z * alpha);
    atomicAdd(pa + q * 4 + 3, v.w * alpha);
  }
}

// ---------- final projection: out = (acc + bias) @ Wp + bp ----------
__global__ void k_proj(const float* __restrict__ acc, const float* __restrict__ bias,
                       const float* __restrict__ Wp, const float* __restrict__ bp,
                       float* __restrict__ out, int nNodes) {
  int t = blockIdx.x * blockDim.x + threadIdx.x;
  int n = t >> 5, j = t & 31;
  if (n >= nNodes) return;
  float s = bp[j];
#pragma unroll 8
  for (int k = 0; k < HC; ++k)
    s = fmaf(acc[(size_t)n * HC + k] + bias[k], Wp[k * DOUT + j], s);
  out[(size_t)n * DOUT + j] = s;
}

extern "C" void kernel_launch(void* const* d_in, const int* in_sizes, int n_in,
                              void* d_out, int out_size, void* d_ws, size_t ws_size,
                              hipStream_t stream) {
  (void)n_in; (void)out_size; (void)ws_size;
  const float* x    = (const float*)d_in[0];
  const int*   edge = (const int*)d_in[1];
  const float* Wl   = (const float*)d_in[2];
  const float* Wr   = (const float*)d_in[3];
  const float* att  = (const float*)d_in[4];
  const float* bias = (const float*)d_in[5];
  const float* Wp   = (const float*)d_in[6];
  const float* bp   = (const float*)d_in[7];

  int nNodes = in_sizes[0] / DIN;
  int nEdges = in_sizes[1] / 2;
  const int* src = edge;
  const int* dsti = edge + nEdges;

  char* ws = (char*)d_ws;
  size_t off = 0;
  auto carve = [&](size_t bytes) {
    void* p = ws + off;
    off += (bytes + 255) & ~(size_t)255;
    return p;
  };
  float*     xl     = (float*)carve((size_t)nNodes * HC * sizeof(float));
  float*     xr     = (float*)carve((size_t)nNodes * HC * sizeof(float));
  float*     acc    = (float*)carve((size_t)nNodes * HC * sizeof(float));
  float*     logits = (float*)carve((size_t)nEdges * HEADS * sizeof(float));
  float*     denom  = (float*)carve((size_t)nNodes * HEADS * sizeof(float));
  unsigned*  mEnc   = (unsigned*)carve((size_t)nNodes * HEADS * sizeof(unsigned));
  _Float16*  WlT    = (_Float16*)carve((size_t)HC * DIN * sizeof(_Float16));
  _Float16*  WrT    = (_Float16*)carve((size_t)HC * DIN * sizeof(_Float16));

  int accN = nNodes * HC, segN = nNodes * HEADS;

  k_prep_weights<<<(HC * DIN + 255) / 256, 256, 0, stream>>>(Wl, Wr, WlT, WrT);
  k_init<<<(accN + 255) / 256, 256, 0, stream>>>(acc, denom, mEnc, accN, segN);
  k_transform<<<(nNodes + 15) / 16, 128, 0, stream>>>(x, WlT, WrT, xl, xr, nNodes);
  k_logits<<<(nEdges + 255) / 256, 256, 0, stream>>>(src, dsti, xl, xr, att, logits, mEnc, nEdges);
  k_expsum<<<(nEdges + 255) / 256, 256, 0, stream>>>(dsti, logits, mEnc, denom, nEdges);
  k_scatter<<<((size_t)nEdges * 4 + 255) / 256, 256, 0, stream>>>(src, dsti, xl, logits, denom, acc, nEdges);
  k_proj<<<((size_t)nNodes * 32 + 255) / 256, 256, 0, stream>>>(acc, bias, Wp, bp, (float*)d_out, nNodes);
}
